// MixtralMoE_45114336477570
// MI455X (gfx1250) — matrix operand order repaired and checked by reading, not compiled, not run on hardware
//
#include <hip/hip_runtime.h>

// ---------------------------------------------------------------------------
// Mixtral MoE (T=4096, H=1024, F=3584, E=8, top-2) for gfx1250 (MI455X).
//  - sparse top-2 routing (4x FLOP cut vs dense reference, same math/token)
//  - bf16 WMMA (v_wmma_f32_16x16x32_bf16), fp32 accum
//  - weights pre-converted fp32->bf16 once (streaming, ~23us HBM) so GEMM
//    inner loops are pure b128 loads + WMMA (no per-element cvt ALU);
//    falls back to in-register conversion if ws is too small.
//  - M_TILE=64 for 64x weight reuse: 16 WMMAs per wave per k-step.
// ---------------------------------------------------------------------------

typedef __attribute__((ext_vector_type(16))) __bf16 v16bf;
typedef __attribute__((ext_vector_type(8)))  __bf16 v8bf;
typedef __attribute__((ext_vector_type(8)))  float  v8f;

#define T_TOK 4096
#define HDIM  1024
#define FDIM  3584
#define NEXP  8
#define MT    64                    // token tile (rows) per block
#define HROWS (8192 + NEXP * MT)    // hbuf rows incl. per-expert padding

__device__ __forceinline__ __bf16 f2bf(float f) {
  unsigned u = __builtin_bit_cast(unsigned, f);
  u += 0x7FFFu + ((u >> 16) & 1u);            // round-to-nearest-even
  unsigned short s = (unsigned short)(u >> 16);
  return __builtin_bit_cast(__bf16, s);
}

// B-operand load: bf16 path = two b128 loads, float path = cvt fallback
__device__ __forceinline__ v16bf loadB(const __bf16* p) {
  return *(const v16bf*)p;
}
__device__ __forceinline__ v16bf loadB(const float* p) {
  v16bf b;
#pragma unroll
  for (int i = 0; i < 16; ++i) b[i] = f2bf(p[i]);
  return b;
}

// ---------------------------------------------------------------- init ------
__global__ void k_init(float* __restrict__ out, int* __restrict__ cnt) {
  size_t i = (size_t)blockIdx.x * blockDim.x + threadIdx.x;
  if (i < (size_t)T_TOK * HDIM) out[i] = 0.0f;
  if (i < NEXP) cnt[i] = 0;
}

// ------------------------------------------------- fp32 -> bf16 stream ------
__global__ void k_cvt(const float* __restrict__ s, __bf16* __restrict__ d, size_t n) {
  size_t i = ((size_t)blockIdx.x * blockDim.x + threadIdx.x) * 4;
  if (i + 3 < n) {
    float4 v = *(const float4*)(s + i);
    d[i] = f2bf(v.x); d[i + 1] = f2bf(v.y); d[i + 2] = f2bf(v.z); d[i + 3] = f2bf(v.w);
  } else {
    for (; i < n; ++i) d[i] = f2bf(s[i]);
  }
}

// -------------------------------------------------------------- router ------
__global__ __launch_bounds__(256) void k_router(
    const float* __restrict__ x, const float* __restrict__ gate_w,
    float* __restrict__ gw, int* __restrict__ cnt, int* __restrict__ idx) {
  const int wave = threadIdx.x >> 5;
  const int lane = threadIdx.x & 31;
  const int t = blockIdx.x * (blockDim.x >> 5) + wave;
  if (t >= T_TOK) return;

  float acc[NEXP];
#pragma unroll
  for (int e = 0; e < NEXP; ++e) acc[e] = 0.0f;

  for (int k = lane; k < HDIM; k += 32) {
    float xv = x[(size_t)t * HDIM + k];
#pragma unroll
    for (int e = 0; e < NEXP; ++e) acc[e] += xv * gate_w[e * HDIM + k];
  }
#pragma unroll
  for (int e = 0; e < NEXP; ++e)
#pragma unroll
    for (int off = 16; off > 0; off >>= 1)
      acc[e] += __shfl_xor(acc[e], off, 32);

  if (lane == 0) {
    int e1 = 0;
#pragma unroll
    for (int e = 1; e < NEXP; ++e) if (acc[e] > acc[e1]) e1 = e;
    int e2 = (e1 == 0) ? 1 : 0;
#pragma unroll
    for (int e = 0; e < NEXP; ++e)
      if (e != e1 && acc[e] > acc[e2]) e2 = e;
    float p2 = __expf(acc[e2] - acc[e1]);      // top-2 softmax, renormalized
    float s  = 1.0f + p2;
    float wa = 1.0f / s, wb = p2 / s;
#pragma unroll
    for (int e = 0; e < NEXP; ++e)
      gw[(size_t)t * NEXP + e] = (e == e1) ? wa : ((e == e2) ? wb : 0.0f);
    int s1 = atomicAdd(&cnt[e1], 1); idx[e1 * T_TOK + s1] = t;
    int s2 = atomicAdd(&cnt[e2], 1); idx[e2 * T_TOK + s2] = t;
  }
}

// ------------------------------------------------------------- offsets ------
__global__ void k_offsets(const int* __restrict__ cnt, int* __restrict__ offs) {
  if (blockIdx.x == 0 && threadIdx.x == 0) {
    int o = 0; offs[0] = 0;
    for (int e = 0; e < NEXP; ++e) { o += (cnt[e] + MT - 1) & ~(MT - 1); offs[e + 1] = o; }
  }
}

// ---------------------------------------------------------------- pass A ----
// h[rows,F] = silu(x @ w1^T) * (x @ w3^T) for gathered tokens of expert e.
// Block: 8 waves, tile 64 tokens x 256 cols; wave: 4 M-subtiles x 2 N-tiles
// x 2 matrices = 16 WMMAs per k-step (K steps of 32 over H).
template <typename TB>
__global__ __launch_bounds__(256) void k_ffn_up(
    const TB* __restrict__ w1, const TB* __restrict__ w3,
    const __bf16* __restrict__ xb, const int* __restrict__ idx,
    const int* __restrict__ cnt, const int* __restrict__ offs,
    __bf16* __restrict__ hbuf) {
  const int e  = blockIdx.z;
  const int m0 = blockIdx.y * MT;
  const int cp = offs[e + 1] - offs[e];
  if (m0 >= cp) return;
  const int cn = cnt[e];

  __shared__ int toks[MT];
  __shared__ __align__(16) __bf16 As[MT][32];

  const int tid = threadIdx.x;
  if (tid < MT) {
    int slot = m0 + tid;
    toks[tid] = idx[e * T_TOK + (slot < cn ? slot : 0)];
  }
  __syncthreads();

  const int wave = tid >> 5;
  const int lane = tid & 31;
  const int hi   = lane >> 4;
  const int l15  = lane & 15;
  const int nb   = blockIdx.x * 256 + wave * 32;

  v8f acc1[4][2], acc3[4][2];
#pragma unroll
  for (int mt = 0; mt < 4; ++mt)
#pragma unroll
    for (int j = 0; j < 2; ++j)
#pragma unroll
      for (int v = 0; v < 8; ++v) { acc1[mt][j][v] = 0.0f; acc3[mt][j][v] = 0.0f; }

  for (int k0 = 0; k0 < HDIM; k0 += 32) {
    // stage gathered A tile (64x32 bf16) in LDS, 8B chunks
#pragma unroll
    for (int j = 0; j < 2; ++j) {
      int q = tid + 256 * j;            // 512 chunks of 4 bf16
      int r = q >> 3, c = (q & 7) * 4;
      *(uint2*)&As[r][c] = *(const uint2*)(xb + (size_t)toks[r] * HDIM + k0 + c);
    }
    __syncthreads();

#pragma unroll
    for (int jj = 0; jj < 2; ++jj) {
      int n = nb + jj * 16 + l15;
      const TB* p1 = w1 + ((size_t)e * FDIM + n) * HDIM + k0 + hi * 16;
      const TB* p3 = w3 + ((size_t)e * FDIM + n) * HDIM + k0 + hi * 16;
      __builtin_prefetch(p1 + 32, 0, 1);
      __builtin_prefetch(p3 + 32, 0, 1);
      v16bf b1 = loadB(p1);
      v16bf b3 = loadB(p3);
#pragma unroll
      for (int mt = 0; mt < 4; ++mt) {
        int m = mt * 16 + l15;
        v8bf alo = *(const v8bf*)&As[m][hi * 16];
        v8bf ahi = *(const v8bf*)&As[m][hi * 16 + 8];
        v16bf a;
#pragma unroll
        for (int i = 0; i < 8; ++i) { a[i] = alo[i]; a[8 + i] = ahi[i]; }
        acc1[mt][jj] = __builtin_amdgcn_wmma_f32_16x16x32_bf16(
            false, a, false, b1, (short)0, acc1[mt][jj], false, false);
        acc3[mt][jj] = __builtin_amdgcn_wmma_f32_16x16x32_bf16(
            false, a, false, b3, (short)0, acc3[mt][jj], false, false);
      }
    }
    __syncthreads();
  }

  const int gbase = offs[e] + m0;
#pragma unroll
  for (int mt = 0; mt < 4; ++mt)
#pragma unroll
    for (int jj = 0; jj < 2; ++jj) {
      int n = nb + jj * 16 + l15;
#pragma unroll
      for (int v = 0; v < 8; ++v) {
        int row = mt * 16 + hi * 8 + v;
        float g = acc1[mt][jj][v];
        float h = (g / (1.0f + __expf(-g))) * acc3[mt][jj][v];   // silu*h3
        hbuf[(size_t)(gbase + row) * FDIM + n] = f2bf(h);
      }
    }
}

// ---------------------------------------------------------------- pass B ----
// out[t] += weight[t,e] * (h @ w2^T); K over F (112 steps of 32).
template <typename TB>
__global__ __launch_bounds__(256) void k_ffn_down(
    const TB* __restrict__ w2, const __bf16* __restrict__ hbuf,
    const float* __restrict__ gw, const int* __restrict__ idx,
    const int* __restrict__ cnt, const int* __restrict__ offs,
    float* __restrict__ out) {
  const int e  = blockIdx.z;
  const int m0 = blockIdx.y * MT;
  const int cp = offs[e + 1] - offs[e];
  if (m0 >= cp) return;
  const int cn = cnt[e];

  __shared__ int   toks[MT];
  __shared__ float wts[MT];
  __shared__ __align__(16) __bf16 As[MT][32];

  const int tid = threadIdx.x;
  if (tid < MT) {
    int slot = m0 + tid;
    if (slot < cn) {
      int t = idx[e * T_TOK + slot];
      toks[tid] = t;
      wts[tid]  = gw[(size_t)t * NEXP + e];
    } else { toks[tid] = 0; wts[tid] = 0.0f; }
  }
  __syncthreads();

  const int wave = tid >> 5;
  const int lane = tid & 31;
  const int hi   = lane >> 4;
  const int l15  = lane & 15;
  const int nb   = blockIdx.x * 256 + wave * 32;
  const int gbase = offs[e] + m0;

  v8f acc[4][2];
#pragma unroll
  for (int mt = 0; mt < 4; ++mt)
#pragma unroll
    for (int j = 0; j < 2; ++j)
#pragma unroll
      for (int v = 0; v < 8; ++v) acc[mt][j][v] = 0.0f;

  for (int k0 = 0; k0 < FDIM; k0 += 32) {
#pragma unroll
    for (int j = 0; j < 2; ++j) {
      int q = tid + 256 * j;
      int r = q >> 3, c = (q & 7) * 4;
      *(uint2*)&As[r][c] = *(const uint2*)(hbuf + (size_t)(gbase + r) * FDIM + k0 + c);
    }
    __syncthreads();

#pragma unroll
    for (int jj = 0; jj < 2; ++jj) {
      int n = nb + jj * 16 + l15;
      const TB* p2 = w2 + ((size_t)e * HDIM + n) * FDIM + k0 + hi * 16;
      __builtin_prefetch(p2 + 32, 0, 1);
      v16bf b = loadB(p2);
#pragma unroll
      for (int mt = 0; mt < 4; ++mt) {
        int m = mt * 16 + l15;
        v8bf alo = *(const v8bf*)&As[m][hi * 16];
        v8bf ahi = *(const v8bf*)&As[m][hi * 16 + 8];
        v16bf a;
#pragma unroll
        for (int i = 0; i < 8; ++i) { a[i] = alo[i]; a[8 + i] = ahi[i]; }
        acc[mt][jj] = __builtin_amdgcn_wmma_f32_16x16x32_bf16(
            false, a, false, b, (short)0, acc[mt][jj], false, false);
      }
    }
    __syncthreads();
  }

#pragma unroll
  for (int mt = 0; mt < 4; ++mt)
#pragma unroll
    for (int jj = 0; jj < 2; ++jj) {
      int n = nb + jj * 16 + l15;
#pragma unroll
      for (int v = 0; v < 8; ++v) {
        int row = mt * 16 + hi * 8 + v;
        float w = wts[row];
        if (w != 0.0f)
          atomicAdd(&out[(size_t)toks[row] * HDIM + n], w * acc[mt][jj][v]);
      }
    }
}

// ---------------------------------------------------------------------------
extern "C" void kernel_launch(void* const* d_in, const int* in_sizes, int n_in,
                              void* d_out, int out_size, void* d_ws, size_t ws_size,
                              hipStream_t stream) {
  (void)in_sizes; (void)n_in; (void)out_size;
  const float* x      = (const float*)d_in[0];   // [T, H]
  const float* gate_w = (const float*)d_in[1];   // [E, H]
  const float* w1     = (const float*)d_in[2];   // [E, F, H]
  const float* w2     = (const float*)d_in[3];   // [E, H, F]
  const float* w3     = (const float*)d_in[4];   // [E, F, H]
  float* out          = (float*)d_out;           // [T, H]

  const size_t szXB  = (size_t)T_TOK * HDIM * 2;          // 8 MB
  const size_t szGW  = (size_t)T_TOK * NEXP * 4;          // 128 KB
  const size_t szIDX = (size_t)NEXP * T_TOK * 4;          // 128 KB
  const size_t szHB  = (size_t)HROWS * FDIM * 2;          // ~62 MB
  const size_t nW    = (size_t)NEXP * FDIM * HDIM;        // elems per weight
  const size_t szW   = nW * 2;                            // ~56 MB each

  char* ws = (char*)d_ws;
  size_t off = 0;
  __bf16* xb  = (__bf16*)(ws + off); off += szXB;
  float*  gw  = (float*) (ws + off); off += szGW;
  int*    cnt = (int*)   (ws + off); off += 64;
  int*    ofs = (int*)   (ws + off); off += 64;
  int*    idx = (int*)   (ws + off); off += szIDX;
  __bf16* hb  = (__bf16*)(ws + off); off += szHB;
  const size_t needBase = off;
  __bf16* w1b = (__bf16*)(ws + off); off += szW;
  __bf16* w3b = (__bf16*)(ws + off); off += szW;
  __bf16* w2b = (__bf16*)(ws + off); off += szW;
  const size_t needFull = off;

  const bool usebf = (ws_size >= needFull) && (ws_size >= needBase);

  const size_t nElem = (size_t)T_TOK * HDIM;
  k_init<<<dim3((unsigned)((nElem + 255) / 256)), dim3(256), 0, stream>>>(out, cnt);
  k_cvt<<<dim3((unsigned)((nElem / 4 + 255) / 256)), dim3(256), 0, stream>>>(x, xb, nElem);
  k_router<<<dim3(T_TOK / 8), dim3(256), 0, stream>>>(x, gate_w, gw, cnt, idx);
  k_offsets<<<dim3(1), dim3(1), 0, stream>>>(cnt, ofs);

  const dim3 gA(FDIM / 256, T_TOK / MT, NEXP);
  const dim3 gB(HDIM / 256, T_TOK / MT, NEXP);
  const unsigned cvtW = (unsigned)((nW / 4 + 255) / 256);

  if (usebf) {
    k_cvt<<<dim3(cvtW), dim3(256), 0, stream>>>(w1, w1b, nW);
    k_cvt<<<dim3(cvtW), dim3(256), 0, stream>>>(w3, w3b, nW);
    k_cvt<<<dim3(cvtW), dim3(256), 0, stream>>>(w2, w2b, nW);
    k_ffn_up<__bf16><<<gA, dim3(256), 0, stream>>>(w1b, w3b, xb, idx, cnt, ofs, hb);
    k_ffn_down<__bf16><<<gB, dim3(256), 0, stream>>>(w2b, hb, gw, idx, cnt, ofs, out);
  } else {
    k_ffn_up<float><<<gA, dim3(256), 0, stream>>>(w1, w3, xb, idx, cnt, ofs, hb);
    k_ffn_down<float><<<gB, dim3(256), 0, stream>>>(w2, hb, gw, idx, cnt, ofs, out);
  }
}
